// CausalSelfAttention_77008763617533
// MI455X (gfx1250) — matrix operand order repaired and checked by reading, not compiled, not run on hardware
//
#include <hip/hip_runtime.h>

typedef __attribute__((ext_vector_type(16))) __bf16 v16bf;
typedef __attribute__((ext_vector_type(8)))  __bf16 v8bf;
typedef __attribute__((ext_vector_type(8)))  float  v8f;
typedef __attribute__((ext_vector_type(4)))  unsigned int v4u;
typedef __attribute__((ext_vector_type(8)))  int    v8i;
typedef __attribute__((ext_vector_type(4)))  int    v4i;

#define DEVINL static __device__ __forceinline__

#if defined(__has_builtin)
# if __has_builtin(__builtin_amdgcn_tensor_load_to_lds) && __has_builtin(__builtin_amdgcn_s_wait_tensorcnt)
#  define HAVE_TDM 1
# endif
#endif

constexpr int   S_LEN = 2048;
constexpr int   CDIM  = 1024;
constexpr int   NHEAD = 16;
constexpr int   HD    = 64;
constexpr int   TOPK  = 128;
constexpr float NEGF  = -3.402823466e38f;

// ---------- helpers ----------
DEVINL unsigned short f2bf(float f) {                 // fp32 -> bf16 RNE
    unsigned u = __float_as_uint(f);
    unsigned r = u + 0x7FFFu + ((u >> 16) & 1u);
    return (unsigned short)(r >> 16);
}
DEVINL unsigned fkey(float f) {                       // order-preserving fp32 -> u32
    unsigned u = __float_as_uint(f);
    return (u & 0x80000000u) ? ~u : (u | 0x80000000u);
}
DEVINL float wred_max(float v) {
    #pragma unroll
    for (int o = 16; o > 0; o >>= 1) v = fmaxf(v, __shfl_xor(v, o, 32));
    return v;
}
DEVINL float wred_sumf(float v) {
    #pragma unroll
    for (int o = 16; o > 0; o >>= 1) v += __shfl_xor(v, o, 32);
    return v;
}
DEVINL int wred_sumi(int v) {
    #pragma unroll
    for (int o = 16; o > 0; o >>= 1) v += __shfl_xor(v, o, 32);
    return v;
}
// two contiguous 16B chunks -> one 16-element bf16 fragment
DEVINL v16bf ld_frag(const unsigned short* lo, const unsigned short* hi) {
    v8bf a = *(const v8bf*)lo;
    v8bf b = *(const v8bf*)hi;
    return __builtin_shufflevector(a, b, 0,1,2,3,4,5,6,7,8,9,10,11,12,13,14,15);
}
DEVINL v8f wmma_bf16(v16bf a, v16bf b, v8f c) {
    return __builtin_amdgcn_wmma_f32_16x16x32_bf16(false, a, false, b, (short)0, c, false, false);
}

#ifdef HAVE_TDM
// Issue one 2D TDM load: tile (tile_d1 rows x tile_d0 bf16) from row-major global
// (row stride = stride_elems) into LDS at lds_off, padding +4 dwords every 16 dwords
// so each 32-element row lands on a 40-element LDS stride.
DEVINL void tdm_load_tile_bf16(unsigned lds_off, const void* gaddr,
                               unsigned tile_d0, unsigned tile_d1,
                               unsigned stride_elems)
{
    unsigned long long ga = (unsigned long long)(size_t)gaddr;
    v4u g0;
    g0[0] = 1u;                                                  // count=1, user D#
    g0[1] = lds_off;                                             // lds_addr
    g0[2] = (unsigned)(ga & 0xFFFFFFFFu);                        // global_addr[31:0]
    g0[3] = (unsigned)((ga >> 32) & 0x01FFFFFFu) | (2u << 30);   // addr[56:32] | type=2
    v8i g1;
    g1[0] = (int)((1u << 16)            // data_size = 1 (2 bytes)
                | (1u << 20)            // pad_enable
                | (3u << 22)            // pad_interval: 16 dwords (one 32-elem row)
                | (3u << 25));          // pad_amount: 4 dwords (8 bf16 -> stride 40)
    g1[1] = (int)((tile_d0 & 0xFFFFu) << 16);                    // tensor_dim0 lo16
    g1[2] = (int)(((tile_d0 >> 16) & 0xFFFFu) | ((tile_d1 & 0xFFFFu) << 16)); // dim0 hi | dim1 lo
    g1[3] = (int)(((tile_d1 >> 16) & 0xFFFFu) | ((tile_d0 & 0xFFFFu) << 16)); // dim1 hi | tile_dim0
    g1[4] = (int)(tile_d1 & 0xFFFFu);                            // tile_dim1 (tile_dim2=0)
    g1[5] = (int)stride_elems;                                   // tensor_dim0_stride lo32
    g1[6] = 0;
    g1[7] = 0;
    v4i z4 = {0, 0, 0, 0};
    v8i z8 = {0, 0, 0, 0, 0, 0, 0, 0};
    __builtin_amdgcn_tensor_load_to_lds(g0, g1, z4, z4, z8, 0);
}
#endif

// ---------- kernel 1: fp32 -> bf16 ----------
__global__ void cvt_bf16(const float* __restrict__ src, unsigned short* __restrict__ dst, int n) {
    int i = blockIdx.x * blockDim.x + threadIdx.x;
    if (i < n) dst[i] = f2bf(src[i]);
}

// ---------- kernel 2: C = A(MxK) * B(NxK)^T, bf16 in, fp32 out ----------
// block = 256 (8 waves), workgroup tile 64(M) x 128(N), wave tile 32x32 (4 WMMA),
// K step 32. TDM double-buffered LDS staging when available.
__global__ __launch_bounds__(256) void gemm_bf16_nt(
    const unsigned short* __restrict__ A,
    const unsigned short* __restrict__ B,
    float* __restrict__ C, int M, int N, int K)
{
    __shared__ unsigned short As[2][64 * 40];
    __shared__ unsigned short Bs[2][128 * 40];
    const int tid  = threadIdx.x;
    const int lane = tid & 31, wave = tid >> 5;
    const int wm = wave >> 2, wn = wave & 3;
    const int m0 = blockIdx.x * 64, n0 = blockIdx.y * 128;
    const int lr = lane & 15;
    const int ak = (lane >> 4) * 16;       // A frag K base (0 / 16)
    const int bk = (lane & 16);            // B frag K base (0 / 16)

    v8f acc00 = {}, acc01 = {}, acc10 = {}, acc11 = {};
    const int nIter = K >> 5;

#ifdef HAVE_TDM
    if (wave == 0) {
        tdm_load_tile_bf16((unsigned)(size_t)(void*)&As[0][0], A + (size_t)m0 * K, 32, 64,  (unsigned)K);
        tdm_load_tile_bf16((unsigned)(size_t)(void*)&Bs[0][0], B + (size_t)n0 * K, 32, 128, (unsigned)K);
    }
    for (int i = 0; i < nIter; ++i) {
        const int cur = i & 1;
        __syncthreads();                         // compute on buf[cur^1] fully drained
        if (wave == 0) {
            if (i + 1 < nIter) {
                const int kk = (i + 1) * 32;
                tdm_load_tile_bf16((unsigned)(size_t)(void*)&As[cur ^ 1][0],
                                   A + (size_t)m0 * K + kk, 32, 64,  (unsigned)K);
                tdm_load_tile_bf16((unsigned)(size_t)(void*)&Bs[cur ^ 1][0],
                                   B + (size_t)n0 * K + kk, 32, 128, (unsigned)K);
                __builtin_amdgcn_s_wait_tensorcnt((short)2);   // pair for step i done
            } else {
                __builtin_amdgcn_s_wait_tensorcnt((short)0);
            }
        }
        __syncthreads();                         // publish buf[cur]
#else
    const int aRow = tid >> 2, aCol = (tid & 3) * 8;   // 64x32 stage, 16B per thread
    const int bRow = tid >> 1, bCol = (tid & 1) * 16;  // 128x32 stage, 32B per thread
    for (int i = 0; i < nIter; ++i) {
        const int cur = 0;
        const int kk = i * 32;
        *(uint4*)(&As[0][aRow * 40 + aCol]) =
            *(const uint4*)(A + (size_t)(m0 + aRow) * K + kk + aCol);
        const unsigned short* gb = B + (size_t)(n0 + bRow) * K + kk + bCol;
        *(uint4*)(&Bs[0][bRow * 40 + bCol])     = *(const uint4*)gb;
        *(uint4*)(&Bs[0][bRow * 40 + bCol + 8]) = *(const uint4*)(gb + 8);
        __syncthreads();
#endif
        const unsigned short* ap0 = &As[cur][(wm * 32 + lr) * 40 + ak];
        const unsigned short* ap1 = &As[cur][(wm * 32 + 16 + lr) * 40 + ak];
        v16bf a0 = ld_frag(ap0, ap0 + 8);
        v16bf a1 = ld_frag(ap1, ap1 + 8);
        const unsigned short* bp0 = &Bs[cur][(wn * 32 + lr) * 40 + bk];
        const unsigned short* bp1 = &Bs[cur][(wn * 32 + 16 + lr) * 40 + bk];
        v16bf b0 = ld_frag(bp0, bp0 + 8);
        v16bf b1 = ld_frag(bp1, bp1 + 8);
        acc00 = wmma_bf16(a0, b0, acc00);
        acc01 = wmma_bf16(a0, b1, acc01);
        acc10 = wmma_bf16(a1, b0, acc10);
        acc11 = wmma_bf16(a1, b1, acc11);
#ifndef HAVE_TDM
        __syncthreads();
#endif
    }
    const int rb = (lane & 16) ? 8 : 0;
    const int row0 = m0 + wm * 32 + rb;
    const int col0 = n0 + wn * 32 + lr;
    #pragma unroll
    for (int r = 0; r < 8; ++r) {
        C[(size_t)(row0 + r) * N + col0]           = acc00[r];
        C[(size_t)(row0 + r) * N + col0 + 16]      = acc01[r];
        C[(size_t)(row0 + 16 + r) * N + col0]      = acc10[r];
        C[(size_t)(row0 + 16 + r) * N + col0 + 16] = acc11[r];
    }
}

// ---------- kernel 3: RoPE + L2 norm, emit bf16 q,k [h][s][64] and vT [h][64][s] ----------
__global__ __launch_bounds__(256) void rope_norm(
    const float* __restrict__ qkv,
    unsigned short* __restrict__ qb, unsigned short* __restrict__ kb,
    unsigned short* __restrict__ vT)
{
    const int t = blockIdx.x * blockDim.x + threadIdx.x;   // token*head
    const int s = t >> 4, h = t & 15;
    const float* qrow = qkv + (size_t)s * (3 * CDIM) + h * HD;
    const float* krow = qrow + CDIM;
    const float* vrow = qrow + 2 * CDIM;
    float rq[HD], rk[HD];
    float nq = 0.f, nk = 0.f;
    #pragma unroll
    for (int i = 0; i < 32; ++i) {
        float ang = (float)s * __powf(10000.0f, -(float)i * (1.0f / 32.0f));
        float sn, cs; __sincosf(ang, &sn, &cs);
        float q0 = qrow[2*i], q1 = qrow[2*i+1];
        rq[2*i]   = q0 * cs - q1 * sn;
        rq[2*i+1] = q1 * cs + q0 * sn;
        nq += rq[2*i]*rq[2*i] + rq[2*i+1]*rq[2*i+1];
        float k0 = krow[2*i], k1 = krow[2*i+1];
        rk[2*i]   = k0 * cs - k1 * sn;
        rk[2*i+1] = k1 * cs + k0 * sn;
        nk += rk[2*i]*rk[2*i] + rk[2*i+1]*rk[2*i+1];
    }
    const float iq = 1.0f / fmaxf(sqrtf(nq), 1e-12f);
    const float ik = 1.0f / fmaxf(sqrtf(nk), 1e-12f);
    const size_t off = ((size_t)h * S_LEN + s) * HD;
    #pragma unroll
    for (int d = 0; d < HD; ++d) {
        qb[off + d] = f2bf(rq[d] * iq);
        kb[off + d] = f2bf(rk[d] * ik);
        vT[((size_t)h * HD + d) * S_LEN + s] = f2bf(vrow[d]);
    }
}

// ---------- kernel 4: causal top-k attention, one (head, 16-query tile) per WG ----------
constexpr int    SCS      = 2056;                       // fp32 row stride (16B aligned)
constexpr size_t PB_OFF   = (size_t)16 * SCS * 4;       // 131584
constexpr size_t RED_OFF  = PB_OFF + (size_t)16 * SCS * 2;   // + 65792
constexpr size_t RS_OFF   = RED_OFF + (size_t)8 * 16 * 64 * 4;
constexpr size_t ATTN_LDS = RS_OFF + 64;                // ~230 KB (< 320 KB/WGP)

__global__ __launch_bounds__(256) void attn_topk(
    const unsigned short* __restrict__ qb,
    const unsigned short* __restrict__ kb,
    const unsigned short* __restrict__ vT,
    unsigned short* __restrict__ yb)
{
    extern __shared__ char smem[];
    float*          sc     = (float*)smem;                       // [16][SCS] fp32 scores
    unsigned short* pb     = (unsigned short*)(smem + PB_OFF);   // [16][SCS] bf16 probs
    float*          red    = (float*)(smem + RED_OFF);           // [8][16][64] partials
    float*          rowsum = (float*)(smem + RS_OFF);            // [16]

    const int qt = blockIdx.x, h = blockIdx.y;
    const int q0 = qt * 16;
    const int lane = threadIdx.x & 31, wave = threadIdx.x >> 5;
    const int lr = lane & 15;
    const int ak = (lane >> 4) * 16;       // A frag K base
    const int bkd = (lane & 16);           // B frag K base
    const int rbase = (lane & 16) ? 8 : 0; // D frag row base

    // ---- phase 1: scores = (q kT)/8, causal masked, into LDS ----
    const unsigned short* qbase = qb + ((size_t)h * S_LEN + q0 + lr) * HD;
    v16bf aq0 = ld_frag(qbase + ak,      qbase + ak + 8);        // dims 0..31
    v16bf aq1 = ld_frag(qbase + 32 + ak, qbase + 32 + ak + 8);   // dims 32..63
    for (int kt = wave; kt <= qt; kt += 8) {
        const int key = kt * 16 + lr;
        const unsigned short* kbase = kb + ((size_t)h * S_LEN + key) * HD;
        v16bf b0 = ld_frag(kbase + bkd,      kbase + bkd + 8);
        v16bf b1 = ld_frag(kbase + 32 + bkd, kbase + 32 + bkd + 8);
        v8f c = {};
        c = wmma_bf16(aq0, b0, c);
        c = wmma_bf16(aq1, b1, c);
        #pragma unroll
        for (int r = 0; r < 8; ++r) {
            const int m = rbase + r;
            float v = c[r] * 0.125f;                 // 1/sqrt(64)
            if (key > q0 + m) v = NEGF;              // causal
            sc[m * SCS + key] = v;
        }
    }
    __syncthreads();

    // ---- phase 2: per-row exact rank-128 threshold + softmax (unnormalized bf16) ----
    const int kmax = (qt + 1) * 16;
    const int kpad = (kmax + 31) & ~31;
    #pragma unroll
    for (int rr = 0; rr < 2; ++rr) {
        const int m = wave * 2 + rr;
        const int L = q0 + m + 1;                    // valid causal length
        const float* row = sc + m * SCS;
        float mx = NEGF;
        for (int k = lane; k < L; k += 32) mx = fmaxf(mx, row[k]);
        mx = wred_max(mx);
        unsigned thr = 0u;
        if (L > TOPK) {                              // bitwise rank select (128th largest)
            for (int bit = 31; bit >= 0; --bit) {
                const unsigned trial = thr | (1u << bit);
                int cnt = 0;
                for (int k = lane; k < L; k += 32) cnt += (fkey(row[k]) >= trial);
                if (wred_sumi(cnt) >= TOPK) thr = trial;
            }
        }
        float sum = 0.f;
        unsigned short* prow = pb + m * SCS;
        for (int k = lane; k < kpad; k += 32) {
            float p = 0.f;
            if (k < L) { const float f = row[k]; if (fkey(f) >= thr) p = __expf(f - mx); }
            sum += p;
            prow[k] = f2bf(p);
        }
        sum = wred_sumf(sum);
        if (lane == 0) rowsum[m] = sum;
    }
    __syncthreads();

    // ---- phase 3: y = P v (K chunks of 32 split across waves) ----
    v8f cy0 = {}, cy1 = {}, cy2 = {}, cy3 = {};
    const int nck = kpad >> 5;
    for (int ck = wave; ck < nck; ck += 8) {
        const unsigned short* pbase = pb + lr * SCS + ck * 32 + ak;
        v16bf ap = ld_frag(pbase, pbase + 8);
        const unsigned short* vbase = vT + ((size_t)h * HD + lr) * S_LEN + ck * 32 + bkd;
        v16bf b0 = ld_frag(vbase,              vbase + 8);
        v16bf b1 = ld_frag(vbase + 16 * S_LEN, vbase + 16 * S_LEN + 8);
        v16bf b2 = ld_frag(vbase + 32 * S_LEN, vbase + 32 * S_LEN + 8);
        v16bf b3 = ld_frag(vbase + 48 * S_LEN, vbase + 48 * S_LEN + 8);
        cy0 = wmma_bf16(ap, b0, cy0);
        cy1 = wmma_bf16(ap, b1, cy1);
        cy2 = wmma_bf16(ap, b2, cy2);
        cy3 = wmma_bf16(ap, b3, cy3);
    }
    #pragma unroll
    for (int r = 0; r < 8; ++r) {
        float* rp = red + ((wave * 16 + rbase + r) * 64) + lr;
        rp[0] = cy0[r]; rp[16] = cy1[r]; rp[32] = cy2[r]; rp[48] = cy3[r];
    }
    __syncthreads();

    // ---- cross-wave reduce, softmax normalize, write y [s][h*64+d] bf16 ----
    const int e0 = threadIdx.x * 4;
    #pragma unroll
    for (int j = 0; j < 4; ++j) {
        const int e = e0 + j, row = e >> 6, col = e & 63;
        float acc = 0.f;
        #pragma unroll
        for (int w = 0; w < 8; ++w) acc += red[(w * 16 + row) * 64 + col];
        acc /= rowsum[row];
        yb[(size_t)(q0 + row) * CDIM + h * HD + col] = f2bf(acc);
    }
}

// ---------- workspace layout (bytes) ----------
constexpr size_t OFF_QKV    = 0;                                   // 2048x3072 fp32
constexpr size_t OFF_XB     = OFF_QKV    + (size_t)S_LEN*3*CDIM*4;
constexpr size_t OFF_WQKVB  = OFF_XB     + (size_t)S_LEN*CDIM*2;
constexpr size_t OFF_WPROJB = OFF_WQKVB  + (size_t)3*CDIM*CDIM*2;
constexpr size_t OFF_QB     = OFF_WPROJB + (size_t)CDIM*CDIM*2;
constexpr size_t OFF_KB     = OFF_QB     + (size_t)NHEAD*S_LEN*HD*2;
constexpr size_t OFF_VT     = OFF_KB     + (size_t)NHEAD*S_LEN*HD*2;
constexpr size_t OFF_YB     = OFF_VT     + (size_t)NHEAD*S_LEN*HD*2;

extern "C" void kernel_launch(void* const* d_in, const int* in_sizes, int n_in,
                              void* d_out, int out_size, void* d_ws, size_t ws_size,
                              hipStream_t stream) {
    const float* x      = (const float*)d_in[0];
    const float* w_qkv  = (const float*)d_in[1];
    const float* w_proj = (const float*)d_in[2];
    float* out = (float*)d_out;
    char*  ws  = (char*)d_ws;

    float*          qkv    = (float*)(ws + OFF_QKV);
    unsigned short* xb     = (unsigned short*)(ws + OFF_XB);
    unsigned short* wqkvb  = (unsigned short*)(ws + OFF_WQKVB);
    unsigned short* wprojb = (unsigned short*)(ws + OFF_WPROJB);
    unsigned short* qbuf   = (unsigned short*)(ws + OFF_QB);
    unsigned short* kbuf   = (unsigned short*)(ws + OFF_KB);
    unsigned short* vtbuf  = (unsigned short*)(ws + OFF_VT);
    unsigned short* ybuf   = (unsigned short*)(ws + OFF_YB);

    cvt_bf16<<<(S_LEN*CDIM + 255) / 256, 256, 0, stream>>>(x, xb, S_LEN*CDIM);
    cvt_bf16<<<(3*CDIM*CDIM + 255) / 256, 256, 0, stream>>>(w_qkv, wqkvb, 3*CDIM*CDIM);
    cvt_bf16<<<(CDIM*CDIM + 255) / 256, 256, 0, stream>>>(w_proj, wprojb, CDIM*CDIM);

    // qkv = x @ w_qkv^T          (2048x1024)x(3072x1024)^T
    gemm_bf16_nt<<<dim3(S_LEN/64, (3*CDIM)/128), 256, 0, stream>>>(xb, wqkvb, qkv,
                                                                   S_LEN, 3*CDIM, CDIM);
    // RoPE + normalize + per-head repack
    rope_norm<<<(S_LEN*NHEAD)/256, 256, 0, stream>>>(qkv, qbuf, kbuf, vtbuf);

    // attention: 128 query tiles x 16 heads
    attn_topk<<<dim3(S_LEN/16, NHEAD), 256, ATTN_LDS, stream>>>(qbuf, kbuf, vtbuf, ybuf);

    // out = y @ w_proj^T         (2048x1024)x(1024x1024)^T
    gemm_bf16_nt<<<dim3(S_LEN/64, CDIM/128), 256, 0, stream>>>(ybuf, wprojb, out,
                                                               S_LEN, CDIM, CDIM);
}
